// Attention_23519240913309
// MI455X (gfx1250) — matrix operand order repaired and checked by reading, not compiled, not run on hardware
//
#include <hip/hip_runtime.h>
#include <math.h>

typedef _Float16 f16;
typedef __attribute__((ext_vector_type(16))) _Float16 v16h;
typedef __attribute__((ext_vector_type(8)))  float    v8f;
typedef __attribute__((ext_vector_type(2)))  _Float16 h2;

#define HEADS   8
#define DK      32
#define DV      64
#define NSEQ    1024
#define DIM     256
#define IDK     256    // HEADS*DK
#define IDV     512    // HEADS*DV
#define QKV_OUT 1024   // IDK + IDK + IDV
#define BATCH   8
#define SCALE_F     0.17677669529663689f   // 32^-0.5
#define INV_SCALE_F 5.656854249492381f     // 32^0.5

// ---------------- WMMA fragment helpers (wave32, 16x16x32 f16) ----------------
// A (16x32, MxK): lane holds row m = lane&15; VGPR j holds K pair:
//   kbase = (j<4?0:16) + (lane>>4)*8 + (j&3)*2
// Source layout: row-major, element (m,k) at base + m*ld + k  (pairs contiguous)
__device__ __forceinline__ v16h load_frag_a(const f16* base, int ld) {
  int lane = threadIdx.x & 31;
  int m = lane & 15, half = lane >> 4;
  v16h a;
#pragma unroll
  for (int j = 0; j < 8; ++j) {
    int kb = ((j < 4) ? 0 : 16) + half * 8 + (j & 3) * 2;
    h2 v = *(const h2*)(base + m * ld + kb);
    a[2 * j] = v[0];
    a[2 * j + 1] = v[1];
  }
  return a;
}

// B (32x16, KxN): lane holds col n = lane&15; VGPR j holds K pair (j<4?0:16) + (lane>>4)*8 + (j&3)*2.
// Source layout: element (k,col) at base + col*ld + k  (K pairs contiguous)
__device__ __forceinline__ v16h load_frag_b(const f16* base, int ld) {
  int lane = threadIdx.x & 31;
  int col = lane & 15, half = lane >> 4;
  v16h b;
#pragma unroll
  for (int j = 0; j < 8; ++j) {
    int k = ((j < 4) ? 0 : 16) + half * 8 + (j & 3) * 2;
    h2 v = *(const h2*)(base + col * ld + k);
    b[2 * j] = v[0];
    b[2 * j + 1] = v[1];
  }
  return b;
}

__device__ __forceinline__ v8f wmma_f16(v16h a, v16h b, v8f c) {
  // (neg_a, A, neg_b, B, c_mod, C, reuse_a, reuse_b)
  return __builtin_amdgcn_wmma_f32_16x16x32_f16(false, a, false, b, (short)0, c,
                                                false, false);
}

// ---------------- prep kernels ----------------
__global__ void prep_params(const float* __restrict__ qg, const float* __restrict__ qb,
                            const float* __restrict__ qm, const float* __restrict__ qv,
                            const float* __restrict__ kg, const float* __restrict__ kb,
                            const float* __restrict__ km, const float* __restrict__ kv,
                            const float* __restrict__ vg, const float* __restrict__ vb,
                            const float* __restrict__ vm, const float* __restrict__ vv,
                            const float* __restrict__ og, const float* __restrict__ ob,
                            const float* __restrict__ om, const float* __restrict__ ov,
                            const float* __restrict__ bo, const float* __restrict__ pos_emb,
                            float* __restrict__ qkv_scale, float* __restrict__ qkv_shift,
                            float* __restrict__ oscale, float* __restrict__ oshift,
                            float* __restrict__ posb) {
  int t = blockIdx.x * blockDim.x + threadIdx.x;
  if (t < QKV_OUT) {
    float g, bb, m, v;
    if (t < IDK)            { g = qg[t];        bb = qb[t];        m = qm[t];        v = qv[t]; }
    else if (t < 2 * IDK)   { int i = t - IDK;   g = kg[i]; bb = kb[i]; m = km[i]; v = kv[i]; }
    else                    { int i = t - 2*IDK; g = vg[i]; bb = vb[i]; m = vm[i]; v = vv[i]; }
    float s = g * rsqrtf(v + 1e-5f);
    qkv_scale[t] = s;
    qkv_shift[t] = bb - m * s;
  }
  if (t < DIM) {
    float s = og[t] * rsqrtf(ov[t] + 1e-5f);
    oscale[t] = s;
    oshift[t] = (bo[t] - om[t]) * s + ob[t];
  }
  if (t < HEADS * NSEQ) {
    int h = t / NSEQ, r = t % NSEQ;
    posb[h * NSEQ + r] = pos_emb[r * HEADS + h] * INV_SCALE_F;
  }
}

__global__ void prep_wqkv(const float* __restrict__ Wq, const float* __restrict__ Wk,
                          const float* __restrict__ Wv, f16* __restrict__ Wqkv16) {
  int t = blockIdx.x * 256 + threadIdx.x;   // [1024][256]
  int o = t >> 8, c = t & 255;
  float w;
  if (o < IDK)          w = Wq[o * DIM + c];
  else if (o < 2 * IDK) w = Wk[(o - IDK) * DIM + c];
  else                  w = Wv[(o - 2 * IDK) * DIM + c];
  Wqkv16[t] = (f16)w;
}

__global__ void prep_wo(const float* __restrict__ Wo, f16* __restrict__ Wo16) {
  int t = blockIdx.x * 256 + threadIdx.x;   // [256][512]
  Wo16[t] = (f16)Wo[t];
}

__global__ void prep_x(const float* __restrict__ x, f16* __restrict__ xT16) {
  int t = blockIdx.x * 256 + threadIdx.x;   // [b][n][c]
  int b = t >> 18;
  int r = t & 262143;
  int n = r >> 8, c = r & 255;
  xT16[t] = (f16)x[(b * DIM + c) * NSEQ + n];
}

// ---------------- QKV projection GEMM + BN + layout scatter ----------------
__global__ void qkv_gemm(const f16* __restrict__ W, const f16* __restrict__ xT,
                         const float* __restrict__ scale, const float* __restrict__ shift,
                         f16* __restrict__ q16, f16* __restrict__ k16,
                         f16* __restrict__ v16) {
  int wave = threadIdx.x >> 5;
  int b = blockIdx.z;
  int obase = blockIdx.y * 16;
  int nbase = (blockIdx.x * 8 + wave) * 16;
  const f16* A = W + obase * DIM;
  const f16* B = xT + (b * NSEQ + nbase) * DIM;
  v8f acc = {};
#pragma unroll
  for (int kk = 0; kk < DIM; kk += 32)
    acc = wmma_f16(load_frag_a(A + kk, DIM), load_frag_b(B + kk, DIM), acc);
  int lane = threadIdx.x & 31;
  int col = lane & 15, half = lane >> 4;
  int n = nbase + col;
#pragma unroll
  for (int j = 0; j < 8; ++j) {
    int o = obase + j + 8 * half;
    float y = acc[j] * scale[o] + shift[o];
    f16 hv = (f16)y;
    if (o < IDK) {                       // Q: [bh][n][dk]
      int h = o >> 5, d = o & 31;
      q16[((b * HEADS + h) * NSEQ + n) * DK + d] = hv;
    } else if (o < 2 * IDK) {            // K: [bh][n][dk]
      int o2 = o - IDK; int h = o2 >> 5, d = o2 & 31;
      k16[((b * HEADS + h) * NSEQ + n) * DK + d] = hv;
    } else {                             // V: [bh][dv][n]
      int o2 = o - 2 * IDK; int h = o2 >> 6, d = o2 & 63;
      v16[((b * HEADS + h) * DV + d) * NSEQ + n] = hv;
    }
  }
}

// ---------------- flash attention + rel-pos bias + GELU ----------------
__global__ void attention(const f16* __restrict__ q16, const f16* __restrict__ k16,
                          const f16* __restrict__ v16, const float* __restrict__ posb,
                          f16* __restrict__ g16) {
  __shared__ __align__(16) f16 Pl[8][16 * 32];   // per-wave P tile (C-layout -> A-layout)
  int wave = threadIdx.x >> 5;
  int lane = threadIdx.x & 31;
  int col = lane & 15, half = lane >> 4;
  int bh = blockIdx.y;
  int b = bh >> 3, h = bh & 7;
  int qbase = (blockIdx.x * 8 + wave) * 16;
  const float* pb = posb + h * NSEQ;

  v16h aq = load_frag_a(q16 + (bh * NSEQ + qbase) * DK, DK);
  float mrow[8], lrow[8];
  v8f o0 = {}, o1 = {}, o2 = {}, o3 = {};
#pragma unroll
  for (int j = 0; j < 8; ++j) { mrow[j] = -1e30f; lrow[j] = 0.f; }
  f16* pl = &Pl[wave][0];

  for (int kt = 0; kt < NSEQ; kt += 32) {
    if (kt + 32 < NSEQ)
      __builtin_prefetch(k16 + (bh * NSEQ + kt + 32) * DK, 0, 0);
    v16h bk0 = load_frag_b(k16 + (bh * NSEQ + kt) * DK, DK);
    v16h bk1 = load_frag_b(k16 + (bh * NSEQ + kt + 16) * DK, DK);
    v8f z = {};
    v8f s0 = wmma_f16(aq, bk0, z);
    v8f s1 = wmma_f16(aq, bk1, z);
#pragma unroll
    for (int j = 0; j < 8; ++j) {
      int i = qbase + j + 8 * half;
      int d0 = i - (kt + col);       if (d0 < 0) d0 = -d0;
      int d1 = i - (kt + 16 + col);  if (d1 < 0) d1 = -d1;
      s0[j] = s0[j] * SCALE_F + pb[d0];
      s1[j] = s1[j] * SCALE_F + pb[d1];
      float mx = fmaxf(s0[j], s1[j]);
#pragma unroll
      for (int off = 1; off < 16; off <<= 1) mx = fmaxf(mx, __shfl_xor(mx, off, 32));
      float mnew = fmaxf(mrow[j], mx);
      float corr = __expf(mrow[j] - mnew);
      mrow[j] = mnew;
      float p0 = __expf(s0[j] - mnew);
      float p1 = __expf(s1[j] - mnew);
      float ps = p0 + p1;
#pragma unroll
      for (int off = 1; off < 16; off <<= 1) ps += __shfl_xor(ps, off, 32);
      lrow[j] = lrow[j] * corr + ps;
      o0[j] *= corr; o1[j] *= corr; o2[j] *= corr; o3[j] *= corr;
      int r = j + 8 * half;                // transpose via LDS: row-major 16x32 P
      pl[r * 32 + col]      = (f16)p0;
      pl[r * 32 + 16 + col] = (f16)p1;
    }
    v16h ap = load_frag_a(pl, 32);         // P in A-layout (K = 32 keys)
    const f16* vbase = v16 + (bh * DV) * NSEQ + kt;   // V: [dv][n], cols = dv
    o0 = wmma_f16(ap, load_frag_b(vbase + 0 * 16 * NSEQ, NSEQ), o0);
    o1 = wmma_f16(ap, load_frag_b(vbase + 1 * 16 * NSEQ, NSEQ), o1);
    o2 = wmma_f16(ap, load_frag_b(vbase + 2 * 16 * NSEQ, NSEQ), o2);
    o3 = wmma_f16(ap, load_frag_b(vbase + 3 * 16 * NSEQ, NSEQ), o3);
  }
  // epilogue: normalize, exact GELU, store g16[b][n][h*DV + d]
#pragma unroll
  for (int j = 0; j < 8; ++j) {
    int qrow = qbase + j + 8 * half;
    float inv = 1.0f / lrow[j];
    f16* gp = g16 + (b * NSEQ + qrow) * IDV + h * DV;
    float y;
    y = o0[j] * inv; y = 0.5f * y * (1.f + erff(y * 0.70710678f)); gp[0 * 16 + col] = (f16)y;
    y = o1[j] * inv; y = 0.5f * y * (1.f + erff(y * 0.70710678f)); gp[1 * 16 + col] = (f16)y;
    y = o2[j] * inv; y = 0.5f * y * (1.f + erff(y * 0.70710678f)); gp[2 * 16 + col] = (f16)y;
    y = o3[j] * inv; y = 0.5f * y * (1.f + erff(y * 0.70710678f)); gp[3 * 16 + col] = (f16)y;
  }
}

// ---------------- output projection + bias + BN ----------------
__global__ void out_proj(const f16* __restrict__ Wo16, const f16* __restrict__ g16,
                         const float* __restrict__ oscale, const float* __restrict__ oshift,
                         float* __restrict__ out) {
  int wave = threadIdx.x >> 5;
  int b = blockIdx.z;
  int obase = blockIdx.y * 16;
  int nbase = (blockIdx.x * 8 + wave) * 16;
  const f16* A = Wo16 + obase * IDV;
  const f16* B = g16 + (b * NSEQ + nbase) * IDV;
  v8f acc = {};
#pragma unroll
  for (int kk = 0; kk < IDV; kk += 32)
    acc = wmma_f16(load_frag_a(A + kk, IDV), load_frag_b(B + kk, IDV), acc);
  int lane = threadIdx.x & 31;
  int col = lane & 15, half = lane >> 4;
  int n = nbase + col;
#pragma unroll
  for (int j = 0; j < 8; ++j) {
    int o = obase + j + 8 * half;
    out[(b * DIM + o) * NSEQ + n] = acc[j] * oscale[o] + oshift[o];
  }
}

// ---------------- launch ----------------
extern "C" void kernel_launch(void* const* d_in, const int* in_sizes, int n_in,
                              void* d_out, int out_size, void* d_ws, size_t ws_size,
                              hipStream_t stream) {
  (void)in_sizes; (void)n_in; (void)out_size; (void)ws_size;
  const float* x       = (const float*)d_in[0];
  const float* Wq      = (const float*)d_in[1];
  const float* Wk      = (const float*)d_in[2];
  const float* Wv      = (const float*)d_in[3];
  const float* Wo      = (const float*)d_in[4];
  const float* bo      = (const float*)d_in[5];
  const float* pos_emb = (const float*)d_in[6];
  const float* qg = (const float*)d_in[7],  *qb = (const float*)d_in[8];
  const float* qm = (const float*)d_in[9],  *qv = (const float*)d_in[10];
  const float* kg = (const float*)d_in[11], *kb = (const float*)d_in[12];
  const float* km = (const float*)d_in[13], *kv = (const float*)d_in[14];
  const float* vg = (const float*)d_in[15], *vb = (const float*)d_in[16];
  const float* vm = (const float*)d_in[17], *vv = (const float*)d_in[18];
  const float* og = (const float*)d_in[19], *ob = (const float*)d_in[20];
  const float* om = (const float*)d_in[21], *ov = (const float*)d_in[22];
  float* out = (float*)d_out;

  char* p = (char*)d_ws;
  auto alloc = [&](size_t bytes) {
    char* r = p;
    p += (bytes + 255) & ~(size_t)255;
    return r;
  };
  f16* xT16    = (f16*)alloc((size_t)BATCH * NSEQ * DIM * 2);
  f16* Wqkv16  = (f16*)alloc((size_t)QKV_OUT * DIM * 2);
  f16* Wo16    = (f16*)alloc((size_t)DIM * IDV * 2);
  f16* q16     = (f16*)alloc((size_t)BATCH * HEADS * NSEQ * DK * 2);
  f16* k16     = (f16*)alloc((size_t)BATCH * HEADS * NSEQ * DK * 2);
  f16* v16     = (f16*)alloc((size_t)BATCH * HEADS * NSEQ * DV * 2);
  f16* g16     = (f16*)alloc((size_t)BATCH * NSEQ * IDV * 2);
  float* qscl  = (float*)alloc(QKV_OUT * 4);
  float* qsft  = (float*)alloc(QKV_OUT * 4);
  float* oscl  = (float*)alloc(DIM * 4);
  float* osft  = (float*)alloc(DIM * 4);
  float* posb  = (float*)alloc(HEADS * NSEQ * 4);

  prep_params<<<32, 256, 0, stream>>>(qg, qb, qm, qv, kg, kb, km, kv,
                                      vg, vb, vm, vv, og, ob, om, ov,
                                      bo, pos_emb, qscl, qsft, oscl, osft, posb);
  prep_wqkv<<<(QKV_OUT * DIM) / 256, 256, 0, stream>>>(Wq, Wk, Wv, Wqkv16);
  prep_wo<<<(DIM * IDV) / 256, 256, 0, stream>>>(Wo, Wo16);
  prep_x<<<(BATCH * NSEQ * DIM) / 256, 256, 0, stream>>>(x, xT16);

  qkv_gemm<<<dim3(8, 64, 8), 256, 0, stream>>>(Wqkv16, xT16, qscl, qsft,
                                               q16, k16, v16);
  attention<<<dim3(8, 64, 1), 256, 0, stream>>>(q16, k16, v16, posb, g16);
  out_proj<<<dim3(8, 16, 8), 256, 0, stream>>>(Wo16, g16, oscl, osft, out);
}
